// Decoder_22393959482018
// MI455X (gfx1250) — compile-verified, matrix operand order checked by reading, not run on hardware
//
#include <hip/hip_runtime.h>
#include <math.h>

// ---------------- types / WMMA helpers ----------------
typedef __attribute__((ext_vector_type(16))) __bf16 bf16x16;
typedef __attribute__((ext_vector_type(8)))  float  f32x8;
typedef __attribute__((ext_vector_type(8)))  unsigned int u32x8;

static constexpr int Bsz = 256;
static constexpr int Ee  = 512;
static constexpr int Hh  = 1024;
static constexpr int Vv  = 128000;
static constexpr int NT1 = Vv - 12;        // 127988
static constexpr int NT1_TILES = 8000;     // ceil(127988/16)

// pack two fp32 -> one dword of two bf16 (round-to-nearest-even)
static __device__ __forceinline__ unsigned pk(float a, float b) {
  unsigned ua = __float_as_uint(a), ub = __float_as_uint(b);
  unsigned ra = (ua + 0x7FFFu + ((ua >> 16) & 1u)) >> 16;
  unsigned rb = (ub + 0x7FFFu + ((ub >> 16) & 1u)) & 0xFFFF0000u;
  return ra | rb;
}

// 16-element bf16 fragment per lane (CDNA5 16-bit A/B layout):
// elements 0..7 = K kb..kb+7, elements 8..15 = K kb+16..kb+23,
// kb = k0 + (lane>>4)*8; lane&15 selects the M row (A) / N column (B).
// fp32 source: load + convert.
static __device__ __forceinline__ u32x8 load_frag(const float* __restrict__ row, int kb) {
  const float4* p0 = reinterpret_cast<const float4*>(row + kb);
  const float4* p1 = reinterpret_cast<const float4*>(row + kb + 16);
  float4 c0 = p0[0], c1 = p0[1];
  float4 c2 = p1[0], c3 = p1[1];
  u32x8 f;
  f[0] = pk(c0.x, c0.y); f[1] = pk(c0.z, c0.w);
  f[2] = pk(c1.x, c1.y); f[3] = pk(c1.z, c1.w);
  f[4] = pk(c2.x, c2.y); f[5] = pk(c2.z, c2.w);
  f[6] = pk(c3.x, c3.y); f[7] = pk(c3.z, c3.w);
  return f;
}

// pre-converted bf16 source: two b128 loads, zero VALU.
static __device__ __forceinline__ u32x8 load_frag(const unsigned short* __restrict__ row, int kb) {
  const uint4* p0 = reinterpret_cast<const uint4*>(row + kb);
  const uint4* p1 = reinterpret_cast<const uint4*>(row + kb + 16);
  uint4 c0 = *p0, c1 = *p1;
  u32x8 f;
  f[0] = c0.x; f[1] = c0.y; f[2] = c0.z; f[3] = c0.w;
  f[4] = c1.x; f[5] = c1.y; f[6] = c1.z; f[7] = c1.w;
  return f;
}

static __device__ __forceinline__ f32x8 wmma_bf16(u32x8 a, u32x8 b, f32x8 c) {
  return __builtin_amdgcn_wmma_f32_16x16x32_bf16(
      false, __builtin_bit_cast(bf16x16, a),
      false, __builtin_bit_cast(bf16x16, b),
      (short)0, c, false, false);
}

static __device__ __forceinline__ float sigf(float x) {
  return 1.0f / (1.0f + __expf(-x));
}

// order-independent (deterministic) float atomic max, nonneg-safe for neg too
static __device__ __forceinline__ void atomicMaxF(float* addr, float v) {
  if (v >= 0.0f) atomicMax((int*)addr, __float_as_int(v));
  else           atomicMin((unsigned int*)addr, (unsigned int)__float_as_int(v));
}

// ---------------- tiny init ----------------
__global__ void k_initb(float* bmax) { *bmax = 0.0f; }

// ---------------- fp32 -> bf16 bulk convert (grid-stride, dword-packed) ----------------
__global__ void k_cvt(const float* __restrict__ in, unsigned int* __restrict__ out, int ndw) {
  int i = blockIdx.x * blockDim.x + threadIdx.x;
  const int stride = gridDim.x * blockDim.x;
  for (; i < ndw; i += stride) {
    float2 v = reinterpret_cast<const float2*>(in)[i];
    out[i] = pk(v.x, v.y);
  }
}

// ---------------- t1_out: (optional) convert + per-matrix max row-norm^2 ----------------
// one wave per row (64 elements); outdw==nullptr -> norms only (small-ws fallback)
__global__ __launch_bounds__(128) void k_t1norm(const float* __restrict__ W,
                                                unsigned int* __restrict__ outdw,
                                                float* __restrict__ bmax) {
  const int lane = threadIdx.x & 31;
  const int r = blockIdx.x * (blockDim.x >> 5) + (threadIdx.x >> 5);
  if (r >= NT1) return;
  float2 v = reinterpret_cast<const float2*>(W + (size_t)r * 64)[lane];
  if (outdw) outdw[(size_t)r * 32 + lane] = pk(v.x, v.y);
  float s = v.x * v.x + v.y * v.y;
  for (int m = 1; m < 32; m <<= 1) s += __shfl_xor(s, m, 32);
  if (lane == 0) atomicMaxF(bmax, s);
}

// ---------------- fused single-step GRU cell (templated on weight storage) ----------------
template <typename WT>
__global__ __launch_bounds__(128) void k_gru_t(
    const float* __restrict__ X, const int* __restrict__ rowidx,
    const float* __restrict__ Hprev,
    const WT* __restrict__ Wih, const WT* __restrict__ Whh,
    const float* __restrict__ bih, const float* __restrict__ bhh,
    float* __restrict__ Hout, int In) {
  const int lane = threadIdx.x & 31;
  const int wave = blockIdx.x * (blockDim.x >> 5) + (threadIdx.x >> 5);
  const int tileM = wave & 15;       // 256/16 row tiles
  const int tileN = wave >> 4;      // 1024/16 col tiles
  const int ln = lane & 15, hl = lane >> 4;

  const int rowA = tileM * 16 + ln;
  const int xr = rowidx ? rowidx[rowA] : rowA;
  const float* arow = X + (size_t)xr * In;
  const float* hrow = Hprev + (size_t)rowA * Hh;
  const int colW = tileN * 16 + ln;   // row of W within one gate block

  const WT* wR = Wih + (size_t)(0 * Hh + colW) * In;
  const WT* wZ = Wih + (size_t)(1 * Hh + colW) * In;
  const WT* wN = Wih + (size_t)(2 * Hh + colW) * In;
  const WT* uR = Whh + (size_t)(0 * Hh + colW) * Hh;
  const WT* uZ = Whh + (size_t)(1 * Hh + colW) * Hh;
  const WT* uN = Whh + (size_t)(2 * Hh + colW) * Hh;

  f32x8 aR = {}, aZ = {}, aN = {}, gR = {}, gZ = {}, gN = {};
  for (int k0 = 0; k0 < In; k0 += 32) {
    int kb = k0 + hl * 8;
    u32x8 a = load_frag(arow, kb);
    aR = wmma_bf16(a, load_frag(wR, kb), aR);
    aZ = wmma_bf16(a, load_frag(wZ, kb), aZ);
    aN = wmma_bf16(a, load_frag(wN, kb), aN);
  }
  for (int k0 = 0; k0 < Hh; k0 += 32) {
    int kb = k0 + hl * 8;
    u32x8 a = load_frag(hrow, kb);
    gR = wmma_bf16(a, load_frag(uR, kb), gR);
    gZ = wmma_bf16(a, load_frag(uZ, kb), gZ);
    gN = wmma_bf16(a, load_frag(uN, kb), gN);
  }

  const int col = tileN * 16 + ln;
  const float bir = bih[col], biz = bih[Hh + col], bin = bih[2 * Hh + col];
  const float bhr = bhh[col], bhz = bhh[Hh + col], bhn = bhh[2 * Hh + col];
#pragma unroll
  for (int q = 0; q < 8; ++q) {
    const int m = tileM * 16 + q + 8 * hl;   // C/D layout: M = vgpr + 8*(lane>=16)
    const float hp = Hprev[(size_t)m * Hh + col];
    const float r  = sigf(aR[q] + bir + gR[q] + bhr);
    const float z  = sigf(aZ[q] + biz + gZ[q] + bhz);
    const float n  = tanhf(aN[q] + bin + r * (gN[q] + bhn));
    Hout[(size_t)m * Hh + col] = (1.0f - z) * n + z * hp;
  }
}

// ---------------- generic NT GEMM: C[256,N] = A[256,K] @ W[N,K]^T ----------------
__global__ __launch_bounds__(128) void k_gemm_nt(
    const float* __restrict__ A, const float* __restrict__ W,
    float* __restrict__ C, int N, int K) {
  const int lane = threadIdx.x & 31;
  const int wave = blockIdx.x * (blockDim.x >> 5) + (threadIdx.x >> 5);
  const int tileM = wave & 15, tileN = wave >> 4;
  const int ln = lane & 15, hl = lane >> 4;
  if (tileN * 16 >= N) return;
  const float* arow = A + (size_t)(tileM * 16 + ln) * K;
  const float* wrow = W + (size_t)(tileN * 16 + ln) * K;
  f32x8 acc = {};
  for (int k0 = 0; k0 < K; k0 += 32) {
    int kb = k0 + hl * 8;
    acc = wmma_bf16(load_frag(arow, kb), load_frag(wrow, kb), acc);
  }
  const int col = tileN * 16 + ln;
#pragma unroll
  for (int q = 0; q < 8; ++q)
    C[(size_t)(tileM * 16 + q + 8 * hl) * N + col] = acc[q];
}

// ---------------- head: log_softmax(out @ head_w^T), 12 cols ----------------
__global__ __launch_bounds__(32) void k_head(
    const float* __restrict__ out, const float* __restrict__ head_w,
    float* __restrict__ head_lp, float* __restrict__ pred) {
  const int b = blockIdx.x, lane = threadIdx.x;
  float s[12];
#pragma unroll
  for (int j = 0; j < 12; ++j) {
    float acc = 0.0f;
    for (int k = lane; k < Hh; k += 32)
      acc += out[(size_t)b * Hh + k] * head_w[(size_t)j * Hh + k];
    for (int m = 1; m < 32; m <<= 1) acc += __shfl_xor(acc, m, 32);
    s[j] = acc;
  }
  if (lane == 0) {
    float mx = s[0];
#pragma unroll
    for (int j = 1; j < 12; ++j) mx = fmaxf(mx, s[j]);
    float se = 0.0f;
#pragma unroll
    for (int j = 0; j < 12; ++j) se += __expf(s[j] - mx);
    const float lse = mx + __logf(se);
#pragma unroll
    for (int j = 0; j < 12; ++j) head_lp[b * 12 + j] = s[j] - lse;
#pragma unroll
    for (int j = 0; j < 10; ++j) pred[(size_t)b * Vv + j] = s[j] - lse;
  }
}

// ---------------- t0 cluster: log_softmax(p0 @ t0_out^T) + head_lp[:,10] ----------------
__global__ __launch_bounds__(256) void k_t0b(
    const float* __restrict__ p0, const float* __restrict__ t0_out,
    const float* __restrict__ head_lp, float* __restrict__ pred) {
  __shared__ float r0[256], r1[256];
  const int b = blockIdx.x, j = threadIdx.x;
  const float p = p0[(size_t)b * 256 + j];
  r0[j] = p * t0_out[j];
  r1[j] = p * t0_out[256 + j];
  __syncthreads();
  for (int s = 128; s > 0; s >>= 1) {
    if (j < s) { r0[j] += r0[j + s]; r1[j] += r1[j + s]; }
    __syncthreads();
  }
  if (j == 0) {
    const float a = r0[0], c = r1[0];
    const float mx = fmaxf(a, c);
    const float lse = mx + __logf(__expf(a - mx) + __expf(c - mx));
    const float hb = head_lp[b * 12 + 10];
    pred[(size_t)b * Vv + 10] = a - lse + hb;
    pred[(size_t)b * Vv + 11] = c - lse + hb;
  }
}

// ---------------- shift[m] = ||out2_m|| * sqrt(max_n ||t1out_n||^2) ----------------
// any per-row upper bound of z keeps exp() <= 1; mathematically identical lse.
__global__ void k_shift(const float* __restrict__ out2, const float* __restrict__ bmax,
                        float* __restrict__ shift) {
  const int r = threadIdx.x;
  float s = 0.0f;
  for (int i = 0; i < 64; ++i) {
    const float v = out2[(size_t)r * 64 + i];
    s += v * v;
  }
  shift[r] = sqrtf(s * (*bmax));
}

// ---------------- t1 pass A: per-(row,slice) partial sum of exp(z - shift) ----------------
template <typename WT>
__global__ __launch_bounds__(128) void k_t1_sum_t(
    const float* __restrict__ A, const WT* __restrict__ W,
    const float* __restrict__ shift, float* __restrict__ partial) {
  const int lane = threadIdx.x & 31, ln = lane & 15, hl = lane >> 4;
  const int wave = blockIdx.x * (blockDim.x >> 5) + (threadIdx.x >> 5); // 512 waves
  const int tileM = wave & 15, slice = wave >> 4;                      // 32 slices
  const float* arow = A + (size_t)(tileM * 16 + ln) * 64;
  const u32x8 a0 = load_frag(arow, hl * 8);
  const u32x8 a1 = load_frag(arow, 32 + hl * 8);
  float sh[8], sm[8];
#pragma unroll
  for (int q = 0; q < 8; ++q) {
    sh[q] = shift[tileM * 16 + q + 8 * hl];
    sm[q] = 0.0f;
  }
  for (int tileN = slice; tileN < NT1_TILES; tileN += 32) {
    const int n = tileN * 16 + ln;
    const bool valid = n < NT1;
    const WT* wrow = W + (size_t)(valid ? n : 0) * 64;
    f32x8 acc = {};
    acc = wmma_bf16(a0, load_frag(wrow, hl * 8), acc);
    acc = wmma_bf16(a1, load_frag(wrow, 32 + hl * 8), acc);
#pragma unroll
    for (int q = 0; q < 8; ++q)
      sm[q] += valid ? __expf(acc[q] - sh[q]) : 0.0f;
  }
#pragma unroll
  for (int q = 0; q < 8; ++q) {
    float v = sm[q];
    for (int s = 1; s < 16; s <<= 1) v += __shfl_xor(v, s, 16);
    if (ln == 0) partial[(size_t)(tileM * 16 + q + 8 * hl) * 32 + slice] = v;
  }
}

// ---------------- finish: lse[row] = shift + log(sum partials), fixed order ----------------
__global__ void k_rowfin(const float* __restrict__ partial,
                         const float* __restrict__ shift,
                         float* __restrict__ lse) {
  const int r = threadIdx.x;
  float s = 0.0f;
  for (int i = 0; i < 32; ++i) s += partial[(size_t)r * 32 + i];
  lse[r] = shift[r] + __logf(s);
}

// ---------------- t1 pass B: prediction[:,12:] = z - lse + head_lp[:,11] ----------------
template <typename WT>
__global__ __launch_bounds__(128) void k_t1_write_t(
    const float* __restrict__ A, const WT* __restrict__ W,
    const float* __restrict__ lse, const float* __restrict__ head_lp,
    float* __restrict__ pred) {
  const int lane = threadIdx.x & 31, ln = lane & 15, hl = lane >> 4;
  const int wave = blockIdx.x * (blockDim.x >> 5) + (threadIdx.x >> 5); // 32000 waves
  const int tileM = wave & 15, nblk = wave >> 4;                       // 2000 * 4 tiles
  const float* arow = A + (size_t)(tileM * 16 + ln) * 64;
  const u32x8 a0 = load_frag(arow, hl * 8);
  const u32x8 a1 = load_frag(arow, 32 + hl * 8);
  float base[8];
#pragma unroll
  for (int q = 0; q < 8; ++q) {
    const int m = tileM * 16 + q + 8 * hl;
    base[q] = head_lp[m * 12 + 11] - lse[m];
  }
  for (int t = 0; t < 4; ++t) {
    const int tileN = nblk * 4 + t;
    const int n = tileN * 16 + ln;
    const bool valid = n < NT1;
    const WT* wrow = W + (size_t)(valid ? n : 0) * 64;
    f32x8 acc = {};
    acc = wmma_bf16(a0, load_frag(wrow, hl * 8), acc);
    acc = wmma_bf16(a1, load_frag(wrow, 32 + hl * 8), acc);
    if (valid) {
#pragma unroll
      for (int q = 0; q < 8; ++q) {
        const int m = tileM * 16 + q + 8 * hl;
        pred[(size_t)m * Vv + 12 + n] = acc[q] + base[q];
      }
    }
  }
}

// ---------------- host orchestration ----------------
extern "C" void kernel_launch(void* const* d_in, const int* in_sizes, int n_in,
                              void* d_out, int out_size, void* d_ws, size_t ws_size,
                              hipStream_t stream) {
  (void)in_sizes; (void)n_in; (void)out_size;
  const int*   input  = (const int*)d_in[0];
  const float* hidden = (const float*)d_in[1];
  const float* emb    = (const float*)d_in[3];
  const float* w_ih0  = (const float*)d_in[4];
  const float* w_hh0  = (const float*)d_in[5];
  const float* b_ih0  = (const float*)d_in[6];
  const float* b_hh0  = (const float*)d_in[7];
  const float* w_ih1  = (const float*)d_in[8];
  const float* w_hh1  = (const float*)d_in[9];
  const float* b_ih1  = (const float*)d_in[10];
  const float* b_hh1  = (const float*)d_in[11];
  const float* head_w = (const float*)d_in[12];
  const float* t0_proj = (const float*)d_in[13];
  const float* t0_out  = (const float*)d_in[14];
  const float* t1_proj = (const float*)d_in[15];
  const float* t1_out  = (const float*)d_in[16];

  float* pred = (float*)d_out;                         // [256, 128000]
  float* h0   = pred + (size_t)Bsz * Vv;               // new_hidden[0]
  float* h1   = h0 + (size_t)Bsz * Hh;                 // new_hidden[1]

  // ---- workspace layout (floats) ----
  float* ws      = (float*)d_ws;
  float* head_lp = ws;                 // 256*12            -> 3072
  float* out2    = ws + 3072;          // 256*64            -> 19456
  float* lse     = ws + 19456;         // 256               -> 19712
  float* shift   = ws + 19712;         // 256               -> 19968
  float* bmax    = ws + 19968;         // 1 (+pad 16)       -> 19984
  float* partial = ws + 19984;         // 256*32            -> 28176
  float* p0      = ws + 28176;         // 256*256           -> 93712
  // bf16 region (starts 16B-aligned)
  unsigned short* bfb   = (unsigned short*)(ws + 93712);
  unsigned short* wih0b = bfb;                               // 3072*512
  unsigned short* whh0b = wih0b + (size_t)3072 * 512;        // 3072*1024
  unsigned short* wih1b = whh0b + (size_t)3072 * 1024;       // 3072*1024
  unsigned short* whh1b = wih1b + (size_t)3072 * 1024;       // 3072*1024
  unsigned short* t1b   = whh1b + (size_t)3072 * 1024;       // NT1*64
  const size_t WS_NEED = (size_t)93712 * 4 +
      ((size_t)3072 * 512 + 3 * (size_t)3072 * 1024 + (size_t)NT1 * 64) * 2;
  const bool big = ws_size >= WS_NEED;

  k_initb<<<1, 1, 0, stream>>>(bmax);

  if (big) {
    // one-time bf16 conversion of all reused weight matrices (L2-resident after)
    k_cvt<<<1024, 256, 0, stream>>>(w_ih0, (unsigned int*)wih0b, 3072 * 512 / 2);
    k_cvt<<<1024, 256, 0, stream>>>(w_hh0, (unsigned int*)whh0b, 3072 * 1024 / 2);
    k_cvt<<<1024, 256, 0, stream>>>(w_ih1, (unsigned int*)wih1b, 3072 * 1024 / 2);
    k_cvt<<<1024, 256, 0, stream>>>(w_hh1, (unsigned int*)whh1b, 3072 * 1024 / 2);
    k_t1norm<<<(NT1 + 3) / 4, 128, 0, stream>>>(t1_out, (unsigned int*)t1b, bmax);
    k_gru_t<unsigned short><<<256, 128, 0, stream>>>(emb, input, hidden,
        wih0b, whh0b, b_ih0, b_hh0, h0, Ee);
    k_gru_t<unsigned short><<<256, 128, 0, stream>>>(h0, nullptr, hidden + (size_t)Bsz * Hh,
        wih1b, whh1b, b_ih1, b_hh1, h1, Hh);
  } else {
    k_t1norm<<<(NT1 + 3) / 4, 128, 0, stream>>>(t1_out, nullptr, bmax);
    k_gru_t<float><<<256, 128, 0, stream>>>(emb, input, hidden,
        w_ih0, w_hh0, b_ih0, b_hh0, h0, Ee);
    k_gru_t<float><<<256, 128, 0, stream>>>(h0, nullptr, hidden + (size_t)Bsz * Hh,
        w_ih1, w_hh1, b_ih1, b_hh1, h1, Hh);
  }

  // head (writes pred[:,0:10] and head_lp)
  k_head<<<256, 32, 0, stream>>>(h1, head_w, head_lp, pred);
  // t0: p0 = h1 @ t0_proj^T (256x256), then 2-way softmax
  k_gemm_nt<<<64, 128, 0, stream>>>(h1, t0_proj, p0, 256, Hh);
  k_t0b<<<256, 256, 0, stream>>>(p0, t0_out, head_lp, pred);
  // t1: out2 = h1 @ t1_proj^T (256x64)
  k_gemm_nt<<<16, 128, 0, stream>>>(h1, t1_proj, out2, 64, Hh);
  k_shift<<<1, 256, 0, stream>>>(out2, bmax, shift);

  // two L2-resident passes over the vocabulary
  if (big) {
    k_t1_sum_t<unsigned short><<<128, 128, 0, stream>>>(out2, t1b, shift, partial);
    k_rowfin<<<1, 256, 0, stream>>>(partial, shift, lse);
    k_t1_write_t<unsigned short><<<8000, 128, 0, stream>>>(out2, t1b, lse, head_lp, pred);
  } else {
    k_t1_sum_t<float><<<128, 128, 0, stream>>>(out2, t1_out, shift, partial);
    k_rowfin<<<1, 256, 0, stream>>>(partial, shift, lse);
    k_t1_write_t<float><<<8000, 128, 0, stream>>>(out2, t1_out, lse, head_lp, pred);
  }
}
